// Naive_Sparse_Attention_83184926589086
// MI455X (gfx1250) — matrix operand order repaired and checked by reading, not compiled, not run on hardware
//
#include <hip/hip_runtime.h>
#include <hip/hip_bf16.h>

// ---------------- static problem config (mirrors the reference) -------------
#define B_ 2
#define S_ 512
#define D_ 1024
#define H_ 16
#define HD_ 64
#define LDQ_ 3072          // qkv row stride (3*D)
#define C_ 31              // compressed kv length
#define J_ 16              // slc blocks
#define TOPK_ 4
#define LSLC_ 32
#define LWIN_ 256
#define KL_ 128            // TOPK*LSLC
#define SCALE_ 0.125f      // 1/sqrt(64)

typedef _Float16 f16;
typedef __attribute__((ext_vector_type(16))) _Float16 v16h;
typedef __attribute__((ext_vector_type(8)))  float    v8f;

// ---------------------------------------------------------------------------
// f32 -> f16 convert (elementwise)
__global__ void cvt_f16_kernel(const float* __restrict__ in, f16* __restrict__ out, int n) {
  int i = blockIdx.x * 256 + threadIdx.x;
  if (i < n) out[i] = (f16)in[i];
}

// f32 [K,N] -> f16 [N,K] transpose for WMMA B fragments (contiguous-K loads)
__global__ void cvt_wt_kernel(const float* __restrict__ W, f16* __restrict__ Wt) {
  int i = blockIdx.x * 256 + threadIdx.x;        // i = n*D_ + k
  if (i >= D_ * LDQ_) return;
  int n = i / D_, k = i - n * D_;
  Wt[i] = (f16)W[(size_t)k * LDQ_ + n];
}

// ---------------------------------------------------------------------------
// QKV GEMM: qkv[1024,3072] = x_h[1024,1024] @ W + b_qkv (W pre-transposed f16)
// One wave computes a 32x64 tile: 2 A frags x 4 B frags -> 8 WMMA per K-step.
__global__ void qkv_gemm_kernel(const f16* __restrict__ A, const f16* __restrict__ Bt,
                                const float* __restrict__ bias, float* __restrict__ Cm) {
  const int lane = threadIdx.x;
  const int m0 = blockIdx.x * 32;
  const int n0 = blockIdx.y * 64;
  const int l15 = lane & 15;
  const int kbA = (lane >> 4) * 16;    // A: lanes16-31 hold K 16..31
  const int kbB = (lane >> 4) * 16;    // B: lanes16-31 hold K 16..31
  v8f acc[8];
#pragma unroll
  for (int i = 0; i < 8; ++i) acc[i] = (v8f){};
  for (int k0 = 0; k0 < D_; k0 += 32) {
    v16h a[2];
#pragma unroll
    for (int i = 0; i < 2; ++i) {
      const f16* arow = A + (size_t)(m0 + i * 16 + l15) * D_ + k0 + kbA;
      __builtin_prefetch(arow + 64, 0, 1);       // K-step after next (global_prefetch_b8)
#pragma unroll
      for (int e = 0; e < 16; ++e) a[i][e] = arow[e];
    }
#pragma unroll
    for (int j = 0; j < 4; ++j) {
      const f16* brow = Bt + (size_t)(n0 + j * 16 + l15) * D_ + k0 + kbB;
      __builtin_prefetch(brow + 64, 0, 1);
      v16h bf;
#pragma unroll
      for (int e = 0; e < 16; ++e) bf[e] = brow[e];
#pragma unroll
      for (int i = 0; i < 2; ++i)
        acc[i * 4 + j] = __builtin_amdgcn_wmma_f32_16x16x32_f16(
            false, a[i], false, bf, (short)0, acc[i * 4 + j], false, false);
    }
  }
#pragma unroll
  for (int i = 0; i < 2; ++i)
#pragma unroll
    for (int j = 0; j < 4; ++j) {
      const int n = n0 + j * 16 + l15;
#pragma unroll
      for (int r = 0; r < 8; ++r) {
        const int m = m0 + i * 16 + r + ((lane >> 4) << 3);
        Cm[(size_t)m * LDQ_ + n] = acc[i * 4 + j][r] + bias[n];
      }
    }
}

// ---------------------------------------------------------------------------
// Pack q/k head-major f16 [bh][s][64] and v transposed f16 [bh][64][s]
__global__ void pack_heads_kernel(const float* __restrict__ qkv, f16* __restrict__ q16,
                                  f16* __restrict__ k16, f16* __restrict__ vT16) {
  int i = blockIdx.x * 256 + threadIdx.x;        // B*H*S*HD = 1,048,576
  if (i >= B_ * H_ * S_ * HD_) return;
  int d = i & 63;
  int s = (i >> 6) & (S_ - 1);
  int bh = i >> 15;
  int b = bh >> 4, h = bh & 15;
  const float* row = qkv + (size_t)(b * S_ + s) * LDQ_ + h * HD_ + d;
  q16[i] = (f16)row[0];
  k16[i] = (f16)row[D_];
  vT16[(size_t)(bh * HD_ + d) * S_ + s] = (f16)row[2 * D_];
}

// ---------------------------------------------------------------------------
// gate[B*S,3] = x @ W_gate + b_gate
__global__ void gate_kernel(const float* __restrict__ x, const float* __restrict__ Wg,
                            const float* __restrict__ bg, float* __restrict__ gate) {
  int i = blockIdx.x * 256 + threadIdx.x;
  if (i >= B_ * S_ * 3) return;
  int row = i / 3, j = i % 3;
  float acc = bg[j];
  for (int d = 0; d < D_; ++d) acc += x[(size_t)row * D_ + d] * Wg[d * 3 + j];
  gate[i] = acc;
}

// ---------------------------------------------------------------------------
// k_cmp/v_cmp [B,C,D]: 32-tap weighted window over k/v
__global__ void compress_kernel(const float* __restrict__ qkv,
                                const float* __restrict__ wk, const float* __restrict__ bk,
                                const float* __restrict__ wv, const float* __restrict__ bv,
                                float* __restrict__ kc, float* __restrict__ vc) {
  int i = blockIdx.x * 256 + threadIdx.x;
  if (i >= B_ * C_ * D_) return;
  int d = i & (D_ - 1);
  int c = (i / D_) % C_;
  int b = i / (D_ * C_);
  float ak = bk[0], av = bv[0];
  for (int l = 0; l < 32; ++l) {
    const float* row = qkv + (size_t)(b * S_ + c * 16 + l) * LDQ_;
    ak += row[D_ + d] * wk[l];
    av += row[2 * D_ + d] * wv[l];
  }
  kc[i] = ak;
  vc[i] = av;
}

// ---------------------------------------------------------------------------
// Compressed attention + block map + top-k (one thread per (b,h,t))
__global__ void cmp_attn_kernel(const float* __restrict__ qkv,
                                const float* __restrict__ kc, const float* __restrict__ vc,
                                float* __restrict__ cmp_out, int* __restrict__ idxbuf) {
  int gid = blockIdx.x * 128 + threadIdx.x;
  if (gid >= B_ * H_ * S_) return;
  int t = gid % S_, h = (gid / S_) % H_, b = gid / (S_ * H_);
  __shared__ float pS[128][32];
  float* p = pS[threadIdx.x];
  const float* qrow = qkv + (size_t)(b * S_ + t) * LDQ_ + h * HD_;
  const int nv = (t < C_) ? t : C_;   // valid: c < t (and c < C)
  float m = -INFINITY;
  for (int c = 0; c < C_; ++c) {
    float s = -INFINITY;
    if (c < nv) {
      const float* kr = kc + (size_t)(b * C_ + c) * D_ + h * HD_;
      float acc = 0.f;
      for (int d = 0; d < HD_; ++d) acc += qrow[d] * kr[d];
      s = acc * SCALE_;
    }
    p[c] = s;
    m = fmaxf(m, s);
  }
  if (!(m > -INFINITY)) m = 0.f;       // fully masked row (t == 0)
  float denom = 0.f;
  for (int c = 0; c < C_; ++c) {
    float e = (c < nv) ? __expf(p[c] - m) : 0.f;
    p[c] = e; denom += e;
  }
  float inv = 1.f / ((denom == 0.f) ? 1.f : denom);
  for (int c = 0; c < C_; ++c) p[c] *= inv;
  // output
  float* orow = cmp_out + (size_t)(b * S_ + t) * D_ + h * HD_;
  for (int d = 0; d < HD_; ++d) {
    float acc = 0.f;
    for (int c = 0; c < nv; ++c) acc += p[c] * vc[(size_t)(b * C_ + c) * D_ + h * HD_ + d];
    orow[d] = acc;
  }
  // p_slc[j] = p[2j] + 2*p[2j+1] + p[2j+2]  (BLOCK_MAP weights 1,2,1)
  float ps[J_];
#pragma unroll
  for (int j = 0; j < J_; ++j) {
    float a = p[2 * j];
    if (2 * j + 1 < C_) a += 2.f * p[2 * j + 1];
    if (2 * j + 2 < C_) a += p[2 * j + 2];
    ps[j] = a;
  }
  // top-4, jax semantics: descending value, ties -> lowest index first
  unsigned used = 0u;
  for (int k = 0; k < TOPK_; ++k) {
    float best = -INFINITY; int bi = 0;
#pragma unroll
    for (int j = 0; j < J_; ++j)
      if (!((used >> j) & 1u) && ps[j] > best) { best = ps[j]; bi = j; }
    used |= 1u << bi;
    idxbuf[(size_t)gid * TOPK_ + k] = bi;
  }
}

// ---------------------------------------------------------------------------
// Selected-block branch: one workgroup per (b,t); LDS-staged 128x64 K/V tiles.
__global__ void cls_attn_kernel(const float* __restrict__ qkv, const int* __restrict__ idxbuf,
                                float* __restrict__ cls_out) {
  const int bt = blockIdx.x;
  const int b = bt / S_, t = bt % S_;
  __shared__ float Ks[KL_ * HD_];   // 32KB, reused for V
  __shared__ float sc[H_ * KL_];    // 8KB
  __shared__ int posS[H_ * TOPK_];
  const int tid = threadIdx.x;      // 256 threads
  if (tid < H_ * TOPK_)
    posS[tid] = idxbuf[(size_t)((b * H_ + tid / TOPK_) * S_ + t) * TOPK_ + (tid % TOPK_)];
  __syncthreads();
  // gather K_slc (sum over heads of per-head gathered blocks)
  for (int e = tid; e < KL_ * HD_; e += 256) {
    int kk = e >> 6, d = e & 63;
    int slot = kk >> 5, off = kk & 31;
    float a = 0.f;
    for (int h = 0; h < H_; ++h) {
      int pos = posS[h * TOPK_ + slot] * LSLC_ + off;
      a += qkv[(size_t)(b * S_ + pos) * LDQ_ + D_ + h * HD_ + d];
    }
    Ks[e] = a;
  }
  __syncthreads();
  // scores [16,128]
  for (int e = tid; e < H_ * KL_; e += 256) {
    int h = e >> 7, kk = e & (KL_ - 1);
    const float* qrow = qkv + (size_t)(b * S_ + t) * LDQ_ + h * HD_;
    float s = 0.f;
    for (int d = 0; d < HD_; ++d) s += qrow[d] * Ks[kk * HD_ + d];
    sc[e] = s * SCALE_;
  }
  __syncthreads();
  // softmax per head
  if (tid < H_) {
    float m = -INFINITY;
    for (int kk = 0; kk < KL_; ++kk) m = fmaxf(m, sc[tid * KL_ + kk]);
    float dsum = 0.f;
    for (int kk = 0; kk < KL_; ++kk) { float e = __expf(sc[tid * KL_ + kk] - m); sc[tid * KL_ + kk] = e; dsum += e; }
    float inv = 1.f / dsum;
    for (int kk = 0; kk < KL_; ++kk) sc[tid * KL_ + kk] *= inv;
  }
  __syncthreads();
  // gather V_slc into Ks
  for (int e = tid; e < KL_ * HD_; e += 256) {
    int kk = e >> 6, d = e & 63;
    int slot = kk >> 5, off = kk & 31;
    float a = 0.f;
    for (int h = 0; h < H_; ++h) {
      int pos = posS[h * TOPK_ + slot] * LSLC_ + off;
      a += qkv[(size_t)(b * S_ + pos) * LDQ_ + 2 * D_ + h * HD_ + d];
    }
    Ks[e] = a;
  }
  __syncthreads();
  // output
  for (int e = tid; e < H_ * HD_; e += 256) {
    int h = e >> 6, d = e & 63;
    float acc = 0.f;
    for (int kk = 0; kk < KL_; ++kk) acc += sc[h * KL_ + kk] * Ks[kk * HD_ + d];
    cls_out[(size_t)(b * S_ + t) * D_ + h * HD_ + d] = acc;
  }
}

// ---------------------------------------------------------------------------
// Window branch step 1: scores via WMMA from packed f16 q/k (contiguous frags)
__global__ void win_scores_kernel(const f16* __restrict__ q16, const f16* __restrict__ k16,
                                  f16* __restrict__ p16) {
  const int s0 = blockIdx.x * 16, t0 = blockIdx.y * 16, bh = blockIdx.z;
  if (s0 > t0 + 15) return;                 // fully above-diagonal
  if (s0 + 15 < t0 - (int)LWIN_) return;    // fully below window
  const int lane = threadIdx.x;
  const f16* qb = q16 + (size_t)bh * S_ * HD_;
  const f16* kb = k16 + (size_t)bh * S_ * HD_;
  const int rA = t0 + (lane & 15);
  const int nB = s0 + (lane & 15);
  const int kbA = (lane >> 4) * 16;
  const int kbB = (lane >> 4) * 16;
  v8f c = {};
#pragma unroll
  for (int d0 = 0; d0 < HD_; d0 += 32) {
    v16h a, bf;
#pragma unroll
    for (int e = 0; e < 16; ++e)
      a[e] = qb[(size_t)rA * HD_ + d0 + kbA + e];
#pragma unroll
    for (int e = 0; e < 16; ++e)
      bf[e] = kb[(size_t)nB * HD_ + d0 + kbB + e];
    c = __builtin_amdgcn_wmma_f32_16x16x32_f16(false, a, false, bf, (short)0, c, false, false);
  }
  f16* prow = p16 + (size_t)bh * S_ * S_;
#pragma unroll
  for (int r = 0; r < 8; ++r) {
    const int t = t0 + r + ((lane >> 4) << 3);
    prow[(size_t)t * S_ + s0 + (lane & 15)] = (f16)(c[r] * SCALE_);
  }
}

// Window step 2: masked softmax per row; zeros written outside [t-256, t]
__global__ void win_softmax_kernel(f16* __restrict__ p16) {
  int gid = blockIdx.x * 256 + threadIdx.x;
  if (gid >= B_ * H_ * S_) return;
  const int t = gid % S_, bh = gid / S_;
  f16* row = p16 + (size_t)bh * S_ * S_ + (size_t)t * S_;
  int lo = t - (int)LWIN_; if (lo < 0) lo = 0;
  float m = -INFINITY;
  for (int s = lo; s <= t; ++s) m = fmaxf(m, (float)row[s]);
  float dsum = 0.f;
  for (int s = lo; s <= t; ++s) dsum += __expf((float)row[s] - m);
  const float inv = 1.f / dsum;
  for (int s = 0; s < lo; ++s) row[s] = (f16)0.f;
  for (int s = lo; s <= t; ++s) row[s] = (f16)(__expf((float)row[s] - m) * inv);
  for (int s = t + 1; s < S_; ++s) row[s] = (f16)0.f;
}

// Window step 3: win_out = P @ V via WMMA, V pre-transposed f16 [bh][d][s]
__global__ void win_pv_kernel(const f16* __restrict__ p16, const f16* __restrict__ vT16,
                              float* __restrict__ win_out) {
  const int d0 = blockIdx.x * 16, t0 = blockIdx.y * 16, bh = blockIdx.z;
  const int b = bh >> 4, h = bh & 15;
  const int lane = threadIdx.x;
  const f16* pbase = p16 + (size_t)bh * S_ * S_;
  const f16* vt = vT16 + (size_t)bh * HD_ * S_;
  const int rA = t0 + (lane & 15);
  const int nB = d0 + (lane & 15);
  const int kbA = (lane >> 4) * 16;
  const int kbB = (lane >> 4) * 16;
  v8f c = {};
  int lo = t0 - (int)LWIN_; if (lo < 0) lo = 0;
  lo &= ~31;
  for (int s0 = lo; s0 <= t0 + 15; s0 += 32) {
    v16h a, bf;
#pragma unroll
    for (int e = 0; e < 16; ++e)
      a[e] = pbase[(size_t)rA * S_ + s0 + kbA + e];
#pragma unroll
    for (int e = 0; e < 16; ++e)
      bf[e] = vt[(size_t)nB * S_ + s0 + kbB + e];
    c = __builtin_amdgcn_wmma_f32_16x16x32_f16(false, a, false, bf, (short)0, c, false, false);
  }
#pragma unroll
  for (int r = 0; r < 8; ++r) {
    const int t = t0 + r + ((lane >> 4) << 3);
    win_out[(size_t)(b * S_ + t) * D_ + h * HD_ + nB] = c[r];
  }
}

// ---------------------------------------------------------------------------
// Gated fusion
__global__ void fuse_kernel(const float* __restrict__ gate, const float* __restrict__ cmp,
                            const float* __restrict__ cls, const float* __restrict__ win,
                            float* __restrict__ out) {
  int i = blockIdx.x * 256 + threadIdx.x;
  if (i >= B_ * S_ * D_) return;
  int row = i >> 10;   // / D_
  out[i] = gate[row * 3 + 0] * cmp[i] + gate[row * 3 + 1] * cls[i] + gate[row * 3 + 2] * win[i];
}

// ---------------------------------------------------------------------------
extern "C" void kernel_launch(void* const* d_in, const int* in_sizes, int n_in,
                              void* d_out, int out_size, void* d_ws, size_t ws_size,
                              hipStream_t stream) {
  (void)in_sizes; (void)n_in; (void)out_size; (void)ws_size;
  const float* x      = (const float*)d_in[0];
  const float* W_qkv  = (const float*)d_in[1];
  const float* b_qkv  = (const float*)d_in[2];
  const float* w_kcmp = (const float*)d_in[3];
  const float* b_kcmp = (const float*)d_in[4];
  const float* w_vcmp = (const float*)d_in[5];
  const float* b_vcmp = (const float*)d_in[6];
  const float* W_gate = (const float*)d_in[7];
  const float* b_gate = (const float*)d_in[8];
  float* out = (float*)d_out;

  char* p = (char*)d_ws;
  auto carve = [&](size_t bytes) -> void* {
    void* r = (void*)p;
    p += (bytes + 255) & ~(size_t)255;
    return r;
  };
  float* qkv     = (float*)carve((size_t)B_ * S_ * LDQ_ * 4);       // 12 MB
  f16*   xh      = (f16*)  carve((size_t)B_ * S_ * D_ * 2);         // 2 MB
  f16*   wt      = (f16*)  carve((size_t)D_ * LDQ_ * 2);            // 6 MB (transposed)
  f16*   q16     = (f16*)  carve((size_t)B_ * H_ * S_ * HD_ * 2);   // 2 MB
  f16*   k16     = (f16*)  carve((size_t)B_ * H_ * S_ * HD_ * 2);   // 2 MB
  f16*   vT16    = (f16*)  carve((size_t)B_ * H_ * S_ * HD_ * 2);   // 2 MB
  float* kc      = (float*)carve((size_t)B_ * C_ * D_ * 4);
  float* vc      = (float*)carve((size_t)B_ * C_ * D_ * 4);
  float* cmp_out = (float*)carve((size_t)B_ * S_ * D_ * 4);
  float* cls_out = (float*)carve((size_t)B_ * S_ * D_ * 4);
  float* win_out = (float*)carve((size_t)B_ * S_ * D_ * 4);
  float* gate    = (float*)carve((size_t)B_ * S_ * 3 * 4);
  int*   idxbuf  = (int*)  carve((size_t)B_ * H_ * S_ * TOPK_ * 4);
  f16*   p16     = (f16*)  carve((size_t)B_ * H_ * S_ * S_ * 2);    // 16 MB

  const int NX = B_ * S_ * D_;          // 1,048,576
  const int NW = D_ * LDQ_;             // 3,145,728
  cvt_f16_kernel<<<(NX + 255) / 256, 256, 0, stream>>>(x, xh, NX);
  cvt_wt_kernel<<<(NW + 255) / 256, 256, 0, stream>>>(W_qkv, wt);
  qkv_gemm_kernel<<<dim3((B_ * S_) / 32, LDQ_ / 64), 32, 0, stream>>>(xh, wt, b_qkv, qkv);
  pack_heads_kernel<<<(NX + 255) / 256, 256, 0, stream>>>(qkv, q16, k16, vT16);
  gate_kernel<<<(B_ * S_ * 3 + 255) / 256, 256, 0, stream>>>(x, W_gate, b_gate, gate);
  compress_kernel<<<(B_ * C_ * D_ + 255) / 256, 256, 0, stream>>>(qkv, w_kcmp, b_kcmp, w_vcmp, b_vcmp, kc, vc);
  cmp_attn_kernel<<<(B_ * H_ * S_) / 128, 128, 0, stream>>>(qkv, kc, vc, cmp_out, idxbuf);
  cls_attn_kernel<<<B_ * S_, 256, 0, stream>>>(qkv, idxbuf, cls_out);
  win_scores_kernel<<<dim3(S_ / 16, S_ / 16, B_ * H_), 32, 0, stream>>>(q16, k16, p16);
  win_softmax_kernel<<<(B_ * H_ * S_ + 255) / 256, 256, 0, stream>>>(p16);
  win_pv_kernel<<<dim3(HD_ / 16, S_ / 16, B_ * H_), 32, 0, stream>>>(p16, vT16, win_out);
  fuse_kernel<<<(NX + 255) / 256, 256, 0, stream>>>(gate, cmp_out, cls_out, win_out, out);
}
